// Physics_Attention_plus_16836271800866
// MI455X (gfx1250) — matrix operand order repaired and checked by reading, not compiled, not run on hardware
//
#include <hip/hip_runtime.h>
#include <hip/hip_bf16.h>
#include <math.h>

// ---------------------------------------------------------------------------
// Physics-Attention pipeline for MI455X (gfx1250, wave32, WMMA).
// Heavy GEMMs: bf16 V_WMMA_F32_16X16X32_BF16, f32 accumulate.
// Round 3: k-loop double-buffered (two independent fragment register sets per
// 64-k step) so WMMA->VALU WAR hazards don't force v_nop insertion; b128
// vector loads, 4-tile A reuse, LDS-staged split-K aggregate, prefetch.
// ---------------------------------------------------------------------------

typedef __bf16 bf16_t;
typedef __attribute__((ext_vector_type(16))) __bf16 v16bf;
typedef __attribute__((ext_vector_type(8)))  __bf16 v8bf;
typedef __attribute__((ext_vector_type(8)))  float  v8f;

#define N_PTS 65536
#define DIMX  256
#define HEADS 8
#define DHEAD 64
#define GSL   64
#define HD    512   // HEADS*DHEAD

__device__ __forceinline__ float gelu_exact(float x) {
    return 0.5f * x * (1.0f + erff(x * 0.70710678118654752440f));
}

// pack 16 f32 (four float4's, in fragment half order) -> v16bf
__device__ __forceinline__ v16bf pack16(float4 a, float4 b, float4 c, float4 d) {
    v16bf f;
    f[0]  = (bf16_t)a.x; f[1]  = (bf16_t)a.y; f[2]  = (bf16_t)a.z; f[3]  = (bf16_t)a.w;
    f[4]  = (bf16_t)b.x; f[5]  = (bf16_t)b.y; f[6]  = (bf16_t)b.z; f[7]  = (bf16_t)b.w;
    f[8]  = (bf16_t)c.x; f[9]  = (bf16_t)c.y; f[10] = (bf16_t)c.z; f[11] = (bf16_t)c.w;
    f[12] = (bf16_t)d.x; f[13] = (bf16_t)d.y; f[14] = (bf16_t)d.z; f[15] = (bf16_t)d.w;
    return f;
}

__device__ __forceinline__ v16bf cat8(v8bf a, v8bf b) {
    v16bf f;
#pragma unroll
    for (int i = 0; i < 8; ++i) { f[i] = a[i]; f[8 + i] = b[i]; }
    return f;
}

// A fragment from contiguous-k row: runs [k0+ks*8..+7], [k0+16+ks*8..+7]
__device__ __forceinline__ v16bf a_frag(const float* Arow, int k0, int ks) {
    const float4* p0 = (const float4*)(Arow + k0 + ks * 8);
    const float4* p1 = (const float4*)(Arow + k0 + 16 + ks * 8);
    return pack16(p0[0], p0[1], p1[0], p1[1]);
}
// B fragment from contiguous-k row: runs [k0+ks*8..+7], [k0+16+ks*8..+7]
__device__ __forceinline__ v16bf b_frag(const float* Brow, int k0, int ks) {
    const float4* p0 = (const float4*)(Brow + k0 + ks * 8);
    const float4* p1 = (const float4*)(Brow + k0 + 16 + ks * 8);
    return pack16(p0[0], p0[1], p1[0], p1[1]);
}

// ---------------------------------------------------------------------------
// GEMM: C = A * B^T (+ per-column bias).  Contiguous K (sAk = sBk = sCn = 1).
// A(m,k) at A[bA*z + m*sAm + k]; B(n,k) at B[bB*z + n*sBn + k];
// C(m,n) at C[bC*z + m*sCm + n].  M%64==0, N%64==0, K%64==0.
// block = 128 (4 waves); each wave computes a 16(M) x 64(N) strip.
// ---------------------------------------------------------------------------
__global__ __launch_bounds__(128)
void wmma_gemm_bias(const float* __restrict__ A, const float* __restrict__ B,
                    const float* __restrict__ bias, float* __restrict__ C,
                    int K,
                    long long sAm, long long sBn, long long sCm,
                    long long bA, long long bB, long long bC,
                    int hasBias) {
    const int lane = threadIdx.x & 31;
    const int wave = threadIdx.x >> 5;
    const int n0 = blockIdx.x * 64;
    const int m0 = (blockIdx.y * 4 + wave) * 16;
    const long long z = blockIdx.z;
    const int r16 = lane & 15;          // M row / N col within tile
    const int ks  = lane >> 4;          // half-select

    const float* Arow = A + bA * z + (long long)(m0 + r16) * sAm;
    const float* Br0  = B + bB * z + (long long)(n0 + r16) * sBn;
    const float* Br1  = Br0 + 16 * sBn;
    const float* Br2  = Br0 + 32 * sBn;
    const float* Br3  = Br0 + 48 * sBn;
    float* Cb = C + bC * z;

    v8f acc0 = {}, acc1 = {}, acc2 = {}, acc3 = {};
    for (int k0 = 0; k0 < K; k0 += 64) {
        // ---- phase 0: fragments in register set X ----
        v16bf afX  = a_frag(Arow, k0, ks);
        v16bf bfX0 = b_frag(Br0, k0, ks);
        v16bf bfX1 = b_frag(Br1, k0, ks);
        v16bf bfX2 = b_frag(Br2, k0, ks);
        v16bf bfX3 = b_frag(Br3, k0, ks);
        if (k0 + 128 < K) __builtin_prefetch(Arow + k0 + 128, 0, 1);
        acc0 = __builtin_amdgcn_wmma_f32_16x16x32_bf16(false, afX, false, bfX0, (short)0, acc0, false, false);
        acc1 = __builtin_amdgcn_wmma_f32_16x16x32_bf16(false, afX, false, bfX1, (short)0, acc1, false, false);
        acc2 = __builtin_amdgcn_wmma_f32_16x16x32_bf16(false, afX, false, bfX2, (short)0, acc2, false, false);
        acc3 = __builtin_amdgcn_wmma_f32_16x16x32_bf16(false, afX, false, bfX3, (short)0, acc3, false, false);

        // ---- phase 1: independent register set Y (no WAR on set X) ----
        v16bf afY  = a_frag(Arow, k0 + 32, ks);
        v16bf bfY0 = b_frag(Br0, k0 + 32, ks);
        v16bf bfY1 = b_frag(Br1, k0 + 32, ks);
        v16bf bfY2 = b_frag(Br2, k0 + 32, ks);
        v16bf bfY3 = b_frag(Br3, k0 + 32, ks);
        acc0 = __builtin_amdgcn_wmma_f32_16x16x32_bf16(false, afY, false, bfY0, (short)0, acc0, false, false);
        acc1 = __builtin_amdgcn_wmma_f32_16x16x32_bf16(false, afY, false, bfY1, (short)0, acc1, false, false);
        acc2 = __builtin_amdgcn_wmma_f32_16x16x32_bf16(false, afY, false, bfY2, (short)0, acc2, false, false);
        acc3 = __builtin_amdgcn_wmma_f32_16x16x32_bf16(false, afY, false, bfY3, (short)0, acc3, false, false);
    }

    const float bv0 = hasBias ? bias[n0 + r16]      : 0.0f;
    const float bv1 = hasBias ? bias[n0 + r16 + 16] : 0.0f;
    const float bv2 = hasBias ? bias[n0 + r16 + 32] : 0.0f;
    const float bv3 = hasBias ? bias[n0 + r16 + 48] : 0.0f;
#pragma unroll
    for (int r = 0; r < 8; ++r) {
        int mg = m0 + r + ks * 8;       // C/D layout: lanes>=16 hold rows M+8
        float* crow = Cb + (long long)mg * sCm + n0 + r16;
        crow[0]  = acc0[r] + bv0;
        crow[16] = acc1[r] + bv1;
        crow[32] = acc2[r] + bv2;
        crow[48] = acc3[r] + bv3;
    }
}

// ---------------------------------------------------------------------------
// LDS fragment readers (bf16, k-contiguous rows of length ldk halves).
// ---------------------------------------------------------------------------
__device__ __forceinline__ v16bf lds_a_frag(const __bf16* base, int ldk,
                                            int row0, int k0, int lane) {
    int mr = lane & 15, ks = lane >> 4;
    const __bf16* p = base + (row0 + mr) * ldk + k0 + ks * 8;
    v8bf r0 = *(const v8bf*)p;
    v8bf r1 = *(const v8bf*)(p + 16);
    return cat8(r0, r1);
}
__device__ __forceinline__ v16bf lds_b_frag(const __bf16* base, int ldk,
                                            int row0, int k0, int lane) {
    int nr = lane & 15, ks = lane >> 4;
    const __bf16* p = base + (row0 + nr) * ldk + k0 + ks * 8;
    v8bf r0 = *(const v8bf*)p;
    v8bf r1 = *(const v8bf*)(p + 16);
    return cat8(r0, r1);
}

// ---------------------------------------------------------------------------
// Split-K aggregate: slice_tokens[h,g,c] += sum_n sw[h,n,g]*xmid[n,h*64+c]
// Block = 256 threads (8 waves); coalesced float4 global reads converted to
// bf16 and staged k-contiguous in LDS (pad 8 halves vs bank conflicts);
// fragments via aligned ds_load_b128; f32 global atomics across split-K.
// ---------------------------------------------------------------------------
#define AGG_KB  64
#define AGG_LDK (AGG_KB + 8)

__global__ __launch_bounds__(256)
void wmma_aggregate(const float* __restrict__ sw, const float* __restrict__ xmid,
                    float* __restrict__ stok, int nPerBlock) {
    __shared__ __bf16 lsw[GSL * AGG_LDK];
    __shared__ __bf16 lxm[DHEAD * AGG_LDK];
    const int tid = threadIdx.x;
    const int lane = tid & 31;
    const int wave = tid >> 5;
    const int nsplit = N_PTS / nPerBlock;
    const int h = blockIdx.x / nsplit;
    const int split = blockIdx.x % nsplit;
    const int nbeg = split * nPerBlock;
    const float* swh = sw + (long long)h * N_PTS * GSL;
    const float* xmh = xmid + h * DHEAD;
    const int m0 = (wave >> 1) * 16;        // g-tile of this wave
    const int c0 = (wave & 1) * 32;         // c-half of this wave

    v8f acc0 = {}, acc1 = {};
    for (int nb = nbeg; nb < nbeg + nPerBlock; nb += AGG_KB) {
        __syncthreads();
#pragma unroll
        for (int it = 0; it < (AGG_KB * 16) / 256; ++it) {
            int idx = tid + it * 256;       // 0..1023
            int nl = idx >> 4;              // local n (k) index
            int g4 = (idx & 15) * 4;        // 4 consecutive g / c columns
            float4 vs = *(const float4*)(swh + (long long)(nb + nl) * GSL + g4);
            lsw[(g4 + 0) * AGG_LDK + nl] = (bf16_t)vs.x;
            lsw[(g4 + 1) * AGG_LDK + nl] = (bf16_t)vs.y;
            lsw[(g4 + 2) * AGG_LDK + nl] = (bf16_t)vs.z;
            lsw[(g4 + 3) * AGG_LDK + nl] = (bf16_t)vs.w;
            float4 vx = *(const float4*)(xmh + (long long)(nb + nl) * HD + g4);
            lxm[(g4 + 0) * AGG_LDK + nl] = (bf16_t)vx.x;
            lxm[(g4 + 1) * AGG_LDK + nl] = (bf16_t)vx.y;
            lxm[(g4 + 2) * AGG_LDK + nl] = (bf16_t)vx.z;
            lxm[(g4 + 3) * AGG_LDK + nl] = (bf16_t)vx.w;
        }
        __syncthreads();
#pragma unroll
        for (int k0 = 0; k0 < AGG_KB; k0 += 32) {
            v16bf a  = lds_a_frag(lsw, AGG_LDK, m0, k0, lane);
            v16bf b0 = lds_b_frag(lxm, AGG_LDK, c0, k0, lane);
            v16bf b1 = lds_b_frag(lxm, AGG_LDK, c0 + 16, k0, lane);
            acc0 = __builtin_amdgcn_wmma_f32_16x16x32_bf16(false, a, false, b0, (short)0, acc0, false, false);
            acc1 = __builtin_amdgcn_wmma_f32_16x16x32_bf16(false, a, false, b1, (short)0, acc1, false, false);
        }
    }
    const int nr = lane & 15, ks = lane >> 4;
    float* dst = stok + h * GSL * DHEAD;
#pragma unroll
    for (int r = 0; r < 8; ++r) {
        int g = m0 + r + ks * 8;
        atomicAdd(&dst[g * DHEAD + c0 + nr],      acc0[r]);
        atomicAdd(&dst[g * DHEAD + c0 + 16 + nr], acc1[r]);
    }
}

// ---------------------------------------------------------------------------
// Fused temp-MLP tail + gumbel softmax + norm accumulation (f32 VALU math).
// ---------------------------------------------------------------------------
__global__ __launch_bounds__(128)
void gumbel_route(const float* __restrict__ t1buf, float* __restrict__ sw,
                  const float* __restrict__ u, const float* __restrict__ Wt2,
                  const float* __restrict__ bt2, const float* __restrict__ biasH,
                  float* __restrict__ norm) {
    __shared__ float zrow[128][GSL + 1];
    __shared__ float nacc[GSL];
    const int tid = threadIdx.x;
    const int h = blockIdx.y;
    const int n = blockIdx.x * 128 + tid;
    if (tid < GSL) nacc[tid] = 0.0f;
    __syncthreads();

    const float* t1 = t1buf + (long long)n * HD + h * DHEAD;
    float acc = 0.0f;
    for (int g = 0; g < GSL; ++g)
        acc += gelu_exact(t1[g]) * Wt2[g];
    float temp = fmaxf(gelu_exact(acc + bt2[0]) + biasH[h], 0.01f);
    float rtemp = 1.0f / temp;

    const long long row = ((long long)h * N_PTS + n) * GSL;
    float zmax = -1e30f;
    for (int g = 0; g < GSL; ++g) {
        float uu = u[row + g];
        float gn = -logf(-logf(uu + 1e-8f) + 1e-8f);
        float zv = (sw[row + g] + gn) * rtemp;
        zrow[tid][g] = zv;
        zmax = fmaxf(zmax, zv);
    }
    float zsum = 0.0f;
    for (int g = 0; g < GSL; ++g) {
        float e = expf(zrow[tid][g] - zmax);
        zrow[tid][g] = e;
        zsum += e;
    }
    float inv = 1.0f / zsum;
    for (int g = 0; g < GSL; ++g) {
        float w = zrow[tid][g] * inv;
        sw[row + g] = w;
        atomicAdd(&nacc[g], w);
    }
    __syncthreads();
    if (tid < GSL) atomicAdd(&norm[h * GSL + tid], nacc[tid]);
}

// ---------------------------------------------------------------------------
// Tiny per-head 64x64 slice attention; writes out_slice TRANSPOSED (d,g).
// ---------------------------------------------------------------------------
__global__ __launch_bounds__(256)
void slice_attn(const float* __restrict__ stok, const float* __restrict__ norm,
                const float* __restrict__ Wq, const float* __restrict__ Wk,
                const float* __restrict__ Wv, float* __restrict__ osT) {
    __shared__ float st[GSL][DHEAD];
    __shared__ float qs[GSL][DHEAD];
    __shared__ float ks_[GSL][DHEAD];
    __shared__ float vs[GSL][DHEAD];
    const int h = blockIdx.x;
    const int tid = threadIdx.x;

    for (int i = tid; i < GSL * DHEAD; i += 256) {
        int g = i >> 6, c = i & 63;
        st[g][c] = stok[h * GSL * DHEAD + i] / (norm[h * GSL + g] + 1e-5f);
    }
    __syncthreads();
    for (int i = tid; i < GSL * DHEAD; i += 256) {
        int g = i >> 6, d = i & 63;
        float aq = 0.f, ak = 0.f, av = 0.f;
        for (int c = 0; c < DHEAD; ++c) {
            float s = st[g][c];
            aq += s * Wq[d * DHEAD + c];
            ak += s * Wk[d * DHEAD + c];
            av += s * Wv[d * DHEAD + c];
        }
        qs[g][d] = aq; ks_[g][d] = ak; vs[g][d] = av;
    }
    __syncthreads();
    if (tid < GSL) {
        const int g = tid;
        float mx = -1e30f;
        for (int j = 0; j < GSL; ++j) {
            float s = 0.f;
            for (int d = 0; d < DHEAD; ++d) s += qs[g][d] * ks_[j][d];
            s *= 0.125f;                 // d^-0.5
            st[g][j] = s;
            mx = fmaxf(mx, s);
        }
        float sum = 0.f;
        for (int j = 0; j < GSL; ++j) {
            float e = expf(st[g][j] - mx);
            st[g][j] = e;
            sum += e;
        }
        float inv = 1.0f / sum;
        for (int d = 0; d < DHEAD; ++d) {
            float o = 0.f;
            for (int j = 0; j < GSL; ++j) o += st[g][j] * vs[j][d];
            osT[h * DHEAD * GSL + d * GSL + g] = o * inv;
        }
    }
}

__global__ void zero_f32(float* __restrict__ p, int n) {
    int i = blockIdx.x * blockDim.x + threadIdx.x;
    if (i < n) p[i] = 0.0f;
}

// ---------------------------------------------------------------------------
extern "C" void kernel_launch(void* const* d_in, const int* in_sizes, int n_in,
                              void* d_out, int out_size, void* d_ws, size_t ws_size,
                              hipStream_t stream) {
    const float* x    = (const float*)d_in[0];
    const float* Wx   = (const float*)d_in[1];
    const float* bx   = (const float*)d_in[2];
    const float* Wt1  = (const float*)d_in[3];
    const float* bt1  = (const float*)d_in[4];
    const float* Wt2  = (const float*)d_in[5];
    const float* bt2  = (const float*)d_in[6];
    const float* biasH= (const float*)d_in[7];
    const float* Ws   = (const float*)d_in[8];
    const float* bs   = (const float*)d_in[9];
    const float* Wq   = (const float*)d_in[10];
    const float* Wk   = (const float*)d_in[11];
    const float* Wv   = (const float*)d_in[12];
    const float* Wout = (const float*)d_in[13];
    const float* bout = (const float*)d_in[14];
    const float* u    = (const float*)d_in[15];
    float* out = (float*)d_out;

    // workspace layout (floats)
    float* ws = (float*)d_ws;
    const long long SZ_NM = (long long)N_PTS * HD;       // 33.5M
    float* xmid = ws;                                    // (N, 512)
    float* sw   = xmid + SZ_NM;                          // (H, N, 64) logits -> weights
    float* ymid = sw + SZ_NM;                            // t1 staging, then scatter out
    float* stok = ymid + SZ_NM;                          // (H, 64, 64)
    float* osT  = stok + HEADS * GSL * DHEAD;            // (H, 64, 64) transposed
    float* norm = osT + HEADS * GSL * DHEAD;             // (H, 64)

    // 0) zero atomic accumulators
    zero_f32<<<dim3((HEADS * GSL * DHEAD + 255) / 256), 256, 0, stream>>>(stok, HEADS * GSL * DHEAD);
    zero_f32<<<dim3((HEADS * GSL + 255) / 256), 256, 0, stream>>>(norm, HEADS * GSL);

    const dim3 blk(128);
    // 1) xmid = x @ Wx^T + bx          (M=65536, N=512, K=256)
    wmma_gemm_bias<<<dim3(HD / 64, N_PTS / 64, 1), blk, 0, stream>>>(
        x, Wx, bx, xmid, DIMX,
        DIMX, DIMX, HD, 0, 0, 0, 1);

    // 2) t1 = xmid_h @ Wt1^T + bt1 -> staged in ymid[n*512 + h*64 + g]
    wmma_gemm_bias<<<dim3(GSL / 64, N_PTS / 64, HEADS), blk, 0, stream>>>(
        xmid, Wt1, bt1, ymid, DHEAD,
        HD, DHEAD, HD, DHEAD, 0, DHEAD, 1);

    // 3) logits = xmid_h @ Ws^T + bs -> sw (H,N,G)
    wmma_gemm_bias<<<dim3(GSL / 64, N_PTS / 64, HEADS), blk, 0, stream>>>(
        xmid, Ws, bs, sw, DHEAD,
        HD, DHEAD, GSL, DHEAD, 0, (long long)N_PTS * GSL, 1);

    // 4) temperature + gumbel softmax + norm
    gumbel_route<<<dim3(N_PTS / 128, HEADS), dim3(128), 0, stream>>>(
        ymid, sw, u, Wt2, bt2, biasH, norm);

    // 5) slice_tokens = sw^T @ xmid  (LDS-staged split-K, 1024 n per block)
    wmma_aggregate<<<dim3(HEADS * (N_PTS / 1024)), dim3(256), 0, stream>>>(
        sw, xmid, stok, 1024);

    // 6) tiny attention among slice tokens -> osT (d-major)
    slice_attn<<<dim3(HEADS), dim3(256), 0, stream>>>(stok, norm, Wq, Wk, Wv, osT);

    // 7) scatter: ymid[n, h*64+d] = sw_h @ osT_h^T   (M=65536, N=64, K=64)
    wmma_gemm_bias<<<dim3(GSL / 64, N_PTS / 64, HEADS), blk, 0, stream>>>(
        sw, osT, nullptr, ymid, GSL,
        GSL, GSL, HD,
        (long long)N_PTS * GSL, (long long)GSL * DHEAD, DHEAD, 0);

    // 8) out = ymid @ Wout^T + bout   (M=65536, N=256, K=512)
    wmma_gemm_bias<<<dim3(DIMX / 64, N_PTS / 64, 1), blk, 0, stream>>>(
        ymid, Wout, bout, out, HD,
        HD, HD, DIMX, 0, 0, 0, 1);
}
